// Encoder_68856915690250
// MI455X (gfx1250) — matrix operand order repaired and checked by reading, not compiled, not run on hardware
//
#include <hip/hip_runtime.h>

// ---------------------------------------------------------------------------
// Types for CDNA5 WMMA (wave32, 16x16x32 bf16 -> f32)
// ---------------------------------------------------------------------------
typedef unsigned short u16;
typedef __attribute__((ext_vector_type(16))) __bf16 v16bf;
typedef __attribute__((ext_vector_type(8)))  __bf16 v8bf;
typedef __attribute__((ext_vector_type(8)))  float  v8f;

union Frag16 { v16bf v; v8bf h[2]; };

struct __align__(16) U16x8 { u16 v[8]; };

__device__ __forceinline__ u16 f2bf(float f) {
    unsigned u = __float_as_uint(f);
    unsigned r = u + 0x7FFFu + ((u >> 16) & 1u);   // round to nearest even
    return (u16)(r >> 16);
}

// ---------------------------------------------------------------------------
// Generic batched bf16 WMMA GEMM: C = alpha * A(MxK) * B(KxN) + bias
//   - A, B in bf16 bits (u16), row-major with leading dims lda/ldb
//   - transB: B element (k,n) at B[n*ldb + k]
//   - batch z -> (b = z/nh, h = z%nh), per-axis element strides for A/B/C
//   - outputs: f32 (Cf) and/or bf16 (Cbf); optional ReLU
// Block: 256 thr (8 waves). Tile 128(M) x 128(N), K-step 32.
// Waves 4(M) x 2(N); each wave 32x64 via 2x4 v_wmma_f32_16x16x32_bf16.
// Fast branch-free b128 staging for fully in-bounds tiles; guarded scalar
// staging only for edge tiles.
// ---------------------------------------------------------------------------
#define BM 128
#define BN 128
#define BK 32
#define LDA_S 40    // halves; 80B row stride (16B multiple)
#define LDB_S 136   // halves; 272B row stride (16B multiple)

__global__ __launch_bounds__(256) void k_gemm(
    const u16* __restrict__ A, const u16* __restrict__ B,
    const float* __restrict__ bias,
    float* __restrict__ Cf, u16* __restrict__ Cbf,
    int M, int N, int Kd, int lda, int ldb, int ldc,
    int nh, long long sAb, long long sAh, long long sBb, long long sBh,
    long long sCb, long long sCh,
    float alpha, int transB, int relu)
{
    __shared__ __align__(16) u16 As[BM * LDA_S];
    __shared__ __align__(16) u16 Bs[BK * LDB_S];

    const int tid = threadIdx.x;
    const int bz = blockIdx.z;
    const int bb = bz / nh, hh = bz % nh;
    A += bb * sAb + hh * sAh;
    B += bb * sBb + hh * sBh;
    const long long coff = bb * sCb + hh * sCh;

    const int m0 = blockIdx.y * BM;
    const int n0 = blockIdx.x * BN;
    const int wave = tid >> 5, lane = tid & 31;
    const int l = lane & 15, hi = lane >> 4;
    const int wm = (wave >> 1) * 32;   // 4 waves along M
    const int wn = (wave & 1) * 64;    // 2 waves along N

    const bool full = (m0 + BM <= M) && (n0 + BN <= N);

    v8f acc[2][4] = {};

    for (int kk = 0; kk < Kd; kk += BK) {
        if (full) {
            // ---- fast path: all loads issued up-front as b128 ----
            U16x8 a[2], b[2];
#pragma unroll
            for (int t = 0; t < 2; ++t) {              // A: 128x32 halves
                int i = tid + t * 256;
                int row = i >> 2, c8 = (i & 3) * 8;
                a[t] = *(const U16x8*)(A + (long long)(m0 + row) * lda + kk + c8);
            }
            if (!transB) {
#pragma unroll
                for (int t = 0; t < 2; ++t) {          // B: 32x128 halves
                    int i = tid + t * 256;
                    int krow = i >> 4, n8 = (i & 15) * 8;
                    b[t] = *(const U16x8*)(B + (long long)(kk + krow) * ldb + n0 + n8);
                }
            } else {
#pragma unroll
                for (int t = 0; t < 2; ++t) {          // B^T: contiguous along K
                    int i = tid + t * 256;
                    int n = i >> 2, kc = (i & 3) * 8;
                    b[t] = *(const U16x8*)(B + (long long)(n0 + n) * ldb + kk + kc);
                }
            }
            // prefetch next A K-tile into near caches (global_prefetch_b8)
            if (kk + BK < Kd)
                __builtin_prefetch(
                    A + (long long)(m0 + (tid >> 1)) * lda + kk + BK + (tid & 1) * 16,
                    0, 3);
#pragma unroll
            for (int t = 0; t < 2; ++t) {
                int i = tid + t * 256;
                int row = i >> 2, c8 = (i & 3) * 8;
                *(U16x8*)&As[row * LDA_S + c8] = a[t];
            }
            if (!transB) {
#pragma unroll
                for (int t = 0; t < 2; ++t) {
                    int i = tid + t * 256;
                    int krow = i >> 4, n8 = (i & 15) * 8;
                    *(U16x8*)&Bs[krow * LDB_S + n8] = b[t];
                }
            } else {
#pragma unroll
                for (int t = 0; t < 2; ++t) {
                    int i = tid + t * 256;
                    int n = i >> 2, kc = (i & 3) * 8;
#pragma unroll
                    for (int j = 0; j < 8; ++j)
                        Bs[(kc + j) * LDB_S + n] = b[t].v[j];
                }
            }
        } else {
            // ---- edge path: guarded scalar staging ----
            for (int i = tid; i < BM * BK; i += 256) {
                int row = i >> 5, c = i & 31;
                int gr = m0 + row;
                As[row * LDA_S + c] =
                    (gr < M) ? A[(long long)gr * lda + kk + c] : (u16)0;
            }
            for (int i = tid; i < BK * BN; i += 256) {
                int k = i >> 7, n = i & 127;
                int gn = n0 + n;
                u16 v = 0;
                if (gn < N)
                    v = transB ? B[(long long)gn * ldb + kk + k]
                               : B[(long long)(kk + k) * ldb + gn];
                Bs[k * LDB_S + n] = v;
            }
        }
        __syncthreads();

        // ---- fragments (ISA 16-bit A / B wave32 layouts) + 8 WMMAs ----
        Frag16 af[2], bf[4];
#pragma unroll
        for (int mt = 0; mt < 2; ++mt) {
            const u16* p = &As[(wm + mt * 16 + l) * LDA_S + hi * 8];
            af[mt].h[0] = *(const v8bf*)(p);        // K = hi*8 .. +7
            af[mt].h[1] = *(const v8bf*)(p + 16);   // K = 16+hi*8 .. +7
        }
#pragma unroll
        for (int nt = 0; nt < 4; ++nt) {
            const u16* p = &Bs[(hi * 8) * LDB_S + wn + nt * 16 + l];
#pragma unroll
            for (int j = 0; j < 8; ++j) {
                bf[nt].v[j]     = __builtin_bit_cast(__bf16, p[j * LDB_S]);
                bf[nt].v[8 + j] = __builtin_bit_cast(__bf16, p[(16 + j) * LDB_S]);
            }
        }
#pragma unroll
        for (int mt = 0; mt < 2; ++mt)
#pragma unroll
            for (int nt = 0; nt < 4; ++nt)
                acc[mt][nt] = __builtin_amdgcn_wmma_f32_16x16x32_bf16(
                    false, af[mt].v, false, bf[nt].v,
                    (short)0, acc[mt][nt], false, false);
        __syncthreads();
    }

    // ---- epilogue: C/D layout lane -> N, VGPR r -> M = hi*8 + r ----
#pragma unroll
    for (int mt = 0; mt < 2; ++mt)
#pragma unroll
        for (int nt = 0; nt < 4; ++nt) {
            int n = n0 + wn + nt * 16 + l;
            if (n >= N) continue;
            float bv = bias ? bias[n] : 0.f;
#pragma unroll
            for (int r = 0; r < 8; ++r) {
                int m = m0 + wm + mt * 16 + hi * 8 + r;
                if (m >= M) continue;
                float f = acc[mt][nt][r] * alpha + bv;
                if (relu) f = fmaxf(f, 0.f);
                long long idx = coff + (long long)m * ldc + n;
                if (Cf)  Cf[idx]  = f;
                if (Cbf) Cbf[idx] = f2bf(f);
            }
        }
}

// ---------------------------------------------------------------------------
// f32 -> bf16 convert (grid-stride)
// ---------------------------------------------------------------------------
__global__ void k_cvt(const float* __restrict__ s, u16* __restrict__ d, long long n) {
    for (long long i = (long long)blockIdx.x * blockDim.x + threadIdx.x; i < n;
         i += (long long)gridDim.x * blockDim.x)
        d[i] = f2bf(s[i]);
}

// elementwise z = a + b, dual output (either may be null)
__global__ void k_add(const float* __restrict__ a, const float* __restrict__ b,
                      float* __restrict__ of, u16* __restrict__ obf, long long n) {
    for (long long i = (long long)blockIdx.x * blockDim.x + threadIdx.x; i < n;
         i += (long long)gridDim.x * blockDim.x) {
        float v = a[i] + b[i];
        if (of)  of[i]  = v;
        if (obf) obf[i] = f2bf(v);
    }
}

// ---------------------------------------------------------------------------
// LayerNorm over D=512: y = LN(A (+B)) * g + beta. One wave32 per row.
// ---------------------------------------------------------------------------
__global__ __launch_bounds__(256) void k_ln(
    const float* __restrict__ A, const float* __restrict__ Badd,
    const float* __restrict__ g, const float* __restrict__ be,
    float* __restrict__ Of, u16* __restrict__ Obf, int rows)
{
    int wave = threadIdx.x >> 5, lane = threadIdx.x & 31;
    int row = blockIdx.x * 8 + wave;
    if (row >= rows) return;
    const float* pa = A + (long long)row * 512;
    const float* pb = Badd ? Badd + (long long)row * 512 : nullptr;
    float x[16], s = 0.f, s2 = 0.f;
#pragma unroll
    for (int j = 0; j < 16; ++j) {
        float v = pa[lane + j * 32];
        if (pb) v += pb[lane + j * 32];
        x[j] = v; s += v; s2 += v * v;
    }
#pragma unroll
    for (int o = 16; o > 0; o >>= 1) { s += __shfl_xor(s, o); s2 += __shfl_xor(s2, o); }
    float mean = s * (1.f / 512.f);
    float var  = s2 * (1.f / 512.f) - mean * mean;
    float rstd = rsqrtf(var + 1e-5f);
#pragma unroll
    for (int j = 0; j < 16; ++j) {
        int c = lane + j * 32;
        float y = (x[j] - mean) * rstd * g[c] + be[c];
        long long idx = (long long)row * 512 + c;
        if (Of)  Of[idx]  = y;
        if (Obf) Obf[idx] = f2bf(y);
    }
}

// ---------------------------------------------------------------------------
// softmax over 50 (row stride 64), write bf16 zero-padded to 64. Wave per row.
// ---------------------------------------------------------------------------
__global__ __launch_bounds__(256) void k_softmax50(
    const float* __restrict__ in, u16* __restrict__ out, int rows)
{
    int wave = threadIdx.x >> 5, lane = threadIdx.x & 31;
    int row = blockIdx.x * 8 + wave;
    if (row >= rows) return;
    const float* p = in + (long long)row * 64;
    float v0 = (lane < 50)      ? p[lane]      : -1e30f;
    float v1 = (lane + 32 < 50) ? p[lane + 32] : -1e30f;
    float mx = fmaxf(v0, v1);
#pragma unroll
    for (int o = 16; o > 0; o >>= 1) mx = fmaxf(mx, __shfl_xor(mx, o));
    float e0 = (lane < 50)      ? __expf(v0 - mx) : 0.f;
    float e1 = (lane + 32 < 50) ? __expf(v1 - mx) : 0.f;
    float s = e0 + e1;
#pragma unroll
    for (int o = 16; o > 0; o >>= 1) s += __shfl_xor(s, o);
    float inv = 1.f / s;
    u16* q = out + (long long)row * 64;
    q[lane]      = f2bf(e0 * inv);
    q[lane + 32] = f2bf(e1 * inv);
}

// softmax over groups of 8 (deformable attention weights); thread per group
__global__ void k_softmax8(const float* __restrict__ in, float* __restrict__ out, int ng) {
    int i = blockIdx.x * blockDim.x + threadIdx.x;
    if (i >= ng) return;
    const float* p = in + (long long)i * 8;
    float mx = -1e30f;
#pragma unroll
    for (int k = 0; k < 8; ++k) mx = fmaxf(mx, p[k]);
    float e[8], s = 0.f;
#pragma unroll
    for (int k = 0; k < 8; ++k) { e[k] = __expf(p[k] - mx); s += e[k]; }
    float inv = 1.f / s;
    float* o = out + (long long)i * 8;
#pragma unroll
    for (int k = 0; k < 8; ++k) o[k] = e[k] * inv;
}

// zero masked rows of val (mask: one byte per token)
__global__ void k_mask(float* __restrict__ val, const unsigned char* __restrict__ mask,
                       long long n) {
    for (long long i = (long long)blockIdx.x * blockDim.x + threadIdx.x; i < n;
         i += (long long)gridDim.x * blockDim.x)
        if (mask[i >> 9]) val[i] = 0.f;
}

// ---------------------------------------------------------------------------
// Deformable sampling: block = (b,q), thread = (head h, channel d).
// out[b,q,h*64+d] = sum_k attw[b,q,h,k] * bilinear(val, ref+off) ; bf16 out.
// ---------------------------------------------------------------------------
__global__ __launch_bounds__(512) void k_deform(
    const float* __restrict__ val, const float* __restrict__ off,
    const float* __restrict__ attw, u16* __restrict__ out)
{
    int b = blockIdx.x >> 10;
    int q = blockIdx.x & 1023;
    int h = threadIdx.x >> 6;
    int d = threadIdx.x & 63;
    float rx = ((q & 31) + 0.5f) * (1.f / 32.f);
    float ry = ((q >> 5) + 0.5f) * (1.f / 32.f);
    long long tok = (long long)b * 1024 + q;
    const float* op = off  + tok * 128 + h * 16;
    const float* wp = attw + tok * 64  + h * 8;
    const float* vb = val  + (long long)b * 1024 * 512 + h * 64 + d;
    float acc = 0.f;
#pragma unroll
    for (int k = 0; k < 8; ++k) {
        float px = fminf(fmaxf(rx + op[k * 2 + 0], 0.f), 1.f) * 31.f;
        float py = fminf(fmaxf(ry + op[k * 2 + 1], 0.f), 1.f) * 31.f;
        float fx = floorf(px), fy = floorf(py);
        int x0 = (int)fx, y0 = (int)fy;
        int x1 = x0 + 1; if (x1 > 31) x1 = 31;
        int y1 = y0 + 1; if (y1 > 31) y1 = 31;
        float wx = px - fx, wy = py - fy;
        float v00 = vb[(long long)(y0 * 32 + x0) * 512];
        float v01 = vb[(long long)(y0 * 32 + x1) * 512];
        float v10 = vb[(long long)(y1 * 32 + x0) * 512];
        float v11 = vb[(long long)(y1 * 32 + x1) * 512];
        float sv = v00 * (1.f - wx) * (1.f - wy) + v01 * wx * (1.f - wy)
                 + v10 * (1.f - wx) * wy + v11 * wx * wy;
        acc += wp[k] * sv;
    }
    out[tok * 512 + h * 64 + d] = f2bf(acc);
}

// ---------------------------------------------------------------------------
// Host orchestration
// ---------------------------------------------------------------------------
extern "C" void kernel_launch(void* const* d_in, const int* in_sizes, int n_in,
                              void* d_out, int out_size, void* d_ws, size_t ws_size,
                              hipStream_t stream) {
    (void)in_sizes; (void)n_in; (void)out_size; (void)ws_size;
    const float* input = (const float*)d_in[0];
    const unsigned char* amask = (const unsigned char*)d_in[1];
    const float* pos  = (const float*)d_in[2];
    const float* box  = (const float*)d_in[3];
    const float* pemb = (const float*)d_in[4];
    auto P = [&](int L, int j) { return (const float*)d_in[5 + L * 32 + j]; };
    // param order per layer dict:
    // 0 Wq 1 bq 2 Wk 3 bk 4 Wv 5 bv 6 Wo 7 bo 8 ln1g 9 ln1b 10 f1W 11 f1b
    // 12 f2W 13 f2b 14 lnfg 15 lnfb 16 Wval 17 bval 18 Woff 19 boff 20 Watt
    // 21 batt 22 Wout 23 bout 24 ln2g 25 ln2b 26 g1W 27 g1b 28 g2W 29 g2b
    // 30 ln3g 31 ln3b

    const long long NTOK = 16LL * 1024;     // 16384 query tokens
    const long long TD   = NTOK * 512;      // 8388608
    const long long NKV  = 16LL * 50;       // 800 kv tokens

    char* wp_ = (char*)d_ws;
    auto alloc = [&](size_t bytes) -> void* {
        void* r = (void*)wp_;
        wp_ += (bytes + 255) & ~(size_t)255;
        return r;
    };
    float* bufX    = (float*)alloc(TD * 4);
    float* xq_f    = (float*)alloc(TD * 4);          // reused as src2_f (ln2 out)
    u16*   xq_bf   = (u16*)  alloc(TD * 2);
    float* srcb_f  = (float*)alloc(TD * 4);
    u16*   srcb_bf = (u16*)  alloc(TD * 2);
    u16*   kv_bf   = (u16*)  alloc(NKV * 512 * 2);
    u16*   Qp_bf   = (u16*)  alloc(TD * 2);
    u16*   Kp_bf   = (u16*)  alloc(NKV * 512 * 2);
    u16*   Vp_bf   = (u16*)  alloc(NKV * 512 * 2);
    float* logits  = (float*)alloc(16LL * 8 * 1024 * 64 * 4);  // also val_f (same size)
    u16*   att_bf  = (u16*)  alloc(16LL * 8 * 1024 * 64 * 2);
    u16*   atto_bf = (u16*)  alloc(TD * 2);
    float* o_f     = (float*)alloc(TD * 4);          // Wo out, later deform-att out
    float* oq_f    = (float*)alloc(TD * 4);
    u16*   oq_bf   = (u16*)  alloc(TD * 2);
    u16*   ff_bf   = (u16*)  alloc(NTOK * 2048 * 2);
    float* ff2_f   = (float*)alloc(TD * 4);
    u16*   q2_bf   = (u16*)  alloc(TD * 2);
    float* off_f   = (float*)alloc(NTOK * 128 * 4);
    float* attlg_f = (float*)alloc(NTOK * 64 * 4);
    float* attw_f  = (float*)alloc(NTOK * 64 * 4);
    u16*   samp_bf = (u16*)  alloc(TD * 2);
    u16*   src2_bf = (u16*)  alloc(TD * 2);
    u16*   wslot   = (u16*)  alloc(512LL * 2048 * 2);
    float* val_f   = logits;                          // lifetime-disjoint reuse
    float* src2_f  = xq_f;                            // lifetime-disjoint reuse

    auto cvtL = [&](const float* s, u16* d, long long n) {
        long long b = (n + 255) / 256; int blocks = b > 4096 ? 4096 : (int)b;
        k_cvt<<<blocks, 256, 0, stream>>>(s, d, n);
    };
    auto gemm = [&](const u16* A, const u16* B, const float* bias,
                    float* Cf, u16* Cbf,
                    int M, int N, int Kd, int lda, int ldb, int ldc,
                    int batch, int nh,
                    long long sAb, long long sAh, long long sBb, long long sBh,
                    long long sCb, long long sCh,
                    float alpha, int transB, int relu) {
        dim3 g((N + BN - 1) / BN, (M + BM - 1) / BM, batch);
        k_gemm<<<g, 256, 0, stream>>>(A, B, bias, Cf, Cbf, M, N, Kd, lda, ldb, ldc,
                                      nh, sAb, sAh, sBb, sBh, sCb, sCh,
                                      alpha, transB, relu);
    };
    auto gemmW = [&](const u16* A, const float* W, const float* bias,
                     float* Cf, u16* Cbf, int M, int N, int Kd, int relu) {
        cvtL(W, wslot, (long long)Kd * N);
        gemm(A, wslot, bias, Cf, Cbf, M, N, Kd, Kd, N, N,
             1, 1, 0, 0, 0, 0, 0, 0, 1.f, 0, relu);
    };

    // ---- layer-invariant precompute ----
    k_add<<<4096, 256, 0, stream>>>(box, pemb, nullptr, kv_bf, NKV * 512);     // kv
    k_add<<<4096, 256, 0, stream>>>(input, pos, srcb_f, srcb_bf, TD);          // src base

    const float* cur = input;
    for (int L = 0; L < 3; ++L) {
        // --- QueryLayer MHA (cross-attention onto 50 box tokens) ---
        k_add<<<4096, 256, 0, stream>>>(cur, pos, xq_f, xq_bf, TD);
        gemmW(xq_bf, P(L, 0), P(L, 1), nullptr, Qp_bf, 16384, 512, 512, 0);
        gemmW(kv_bf, P(L, 2), P(L, 3), nullptr, Kp_bf, 800, 512, 512, 0);
        gemmW(kv_bf, P(L, 4), P(L, 5), nullptr, Vp_bf, 800, 512, 512, 0);
        // logits[b,h,q,k'] = Q . K^T / 8 ; batched over 16*8
        gemm(Qp_bf, Kp_bf, nullptr, logits, nullptr,
             1024, 50, 64, 512, 512, 64,
             128, 8, 1024LL * 512, 64, 50LL * 512, 64, 8LL * 1024 * 64, 1024LL * 64,
             0.125f, 1, 0);
        k_softmax50<<<16384, 256, 0, stream>>>(logits, att_bf, 131072);
        // o[b,q,h,d] = att @ V (K padded 50->64 with zeros)
        gemm(att_bf, Vp_bf, nullptr, nullptr, atto_bf,
             1024, 64, 64, 64, 512, 512,
             128, 8, 8LL * 1024 * 64, 1024LL * 64, 50LL * 512, 64, 1024LL * 512, 64,
             1.f, 0, 0);
        gemmW(atto_bf, P(L, 6), P(L, 7), o_f, nullptr, 16384, 512, 512, 0);
        k_ln<<<2048, 256, 0, stream>>>(xq_f, o_f, P(L, 8), P(L, 9), oq_f, oq_bf, 16384);
        // --- query FFN ---
        gemmW(oq_bf, P(L, 10), P(L, 11), nullptr, ff_bf, 16384, 2048, 512, 1);
        gemmW(ff_bf, P(L, 12), P(L, 13), ff2_f, nullptr, 16384, 512, 2048, 0);
        k_ln<<<2048, 256, 0, stream>>>(oq_f, ff2_f, P(L, 14), P(L, 15),
                                       nullptr, q2_bf, 16384);
        // --- deformable attention ---
        gemmW(srcb_bf, P(L, 16), P(L, 17), val_f, nullptr, 16384, 512, 512, 0);
        k_mask<<<4096, 256, 0, stream>>>(val_f, amask, TD);
        gemmW(q2_bf, P(L, 18), P(L, 19), off_f, nullptr, 16384, 128, 512, 0);
        gemmW(q2_bf, P(L, 20), P(L, 21), attlg_f, nullptr, 16384, 64, 512, 0);
        k_softmax8<<<512, 256, 0, stream>>>(attlg_f, attw_f, 131072);
        k_deform<<<16384, 512, 0, stream>>>(val_f, off_f, attw_f, samp_bf);
        gemmW(samp_bf, P(L, 22), P(L, 23), o_f, nullptr, 16384, 512, 512, 0);
        k_ln<<<2048, 256, 0, stream>>>(srcb_f, o_f, P(L, 24), P(L, 25),
                                       src2_f, src2_bf, 16384);
        // --- src FFN ---
        gemmW(src2_bf, P(L, 26), P(L, 27), nullptr, ff_bf, 16384, 2048, 512, 1);
        gemmW(ff_bf, P(L, 28), P(L, 29), ff2_f, nullptr, 16384, 512, 2048, 0);
        float* outp = (L == 2) ? (float*)d_out : bufX;
        k_ln<<<2048, 256, 0, stream>>>(src2_f, ff2_f, P(L, 30), P(L, 31),
                                       outp, nullptr, 16384);
        cur = outp;
    }
}
